// QuantumAttention_65481071405008
// MI455X (gfx1250) — matrix operand order repaired and checked by reading, not compiled, not run on hardware
//
#include <hip/hip_runtime.h>
#include <stdint.h>

// ---------------------------------------------------------------------------
// Problem constants: B=2, S=2048, E=1024, H=16, dk=64, NQ=4
// ---------------------------------------------------------------------------
#define BB 2
#define SS 2048
#define EE 1024
#define HH 16
#define DKK 64
#define MROWS (BB * SS)          // 4096

typedef __attribute__((ext_vector_type(16))) _Float16 v16h;
typedef __attribute__((ext_vector_type(8)))  float    v8f;
typedef __attribute__((ext_vector_type(4)))  int      v4i;

union F16Frag { v16h h; uint32_t u[8]; uint4 q[2]; _Float16 e[16]; };

// A-fragment K index pattern (16-bit A, 16x32): VGPR i holds K pair starting at:
__device__ __forceinline__ int akb(int i, int hf) {
    return ((i < 4) ? (2 * i) : (16 + 2 * (i - 4))) + 8 * hf;
}

#ifndef __has_builtin
#define __has_builtin(x) 0
#endif

// ---------------------------------------------------------------------------
// Async global->LDS staging (gfx1250 ASYNCcnt path), guarded with fallback.
// Builtin signature (from probe diagnostic): param0 = int4 AS(1)* (global src).
// ---------------------------------------------------------------------------
#if __has_builtin(__builtin_amdgcn_global_load_async_to_lds_b128)
#define HAVE_ASYNC_LDS 1
#endif

__device__ __forceinline__ void copy16_async(void* lds, const void* glob) {
#ifdef HAVE_ASYNC_LDS
    __builtin_amdgcn_global_load_async_to_lds_b128(
        (__attribute__((address_space(1))) v4i*)(uintptr_t)glob,
        (__attribute__((address_space(3))) v4i*)(uint32_t)(uintptr_t)lds,
        0, 0);
#else
    *(uint4*)lds = *(const uint4*)glob;
#endif
}

#ifdef HAVE_ASYNC_LDS
#if __has_builtin(__builtin_amdgcn_s_wait_asynccnt)
#define WAIT_ASYNC(n) __builtin_amdgcn_s_wait_asynccnt(n)
#else
#define WAIT_ASYNC(n) asm volatile("s_wait_asynccnt " #n ::: "memory")
#endif
#else
#define WAIT_ASYNC(n) do {} while (0)
#endif

// ---------------------------------------------------------------------------
// Row reductions within 16-lane halves: v_permlane16_b32 (VALU, no DS waits).
// ---------------------------------------------------------------------------
#if __has_builtin(__builtin_amdgcn_permlane16)
__device__ __forceinline__ float pl16(float x, unsigned s0, unsigned s1) {
    unsigned xi = (unsigned)__float_as_int(x);
    unsigned r = __builtin_amdgcn_permlane16(xi, xi, s0, s1, false, false);
    return __int_as_float((int)r);
}
__device__ __forceinline__ float rowmax16(float x) {
    x = fmaxf(x, pl16(x, 0x67452301u, 0xEFCDAB89u));  // xor 1
    x = fmaxf(x, pl16(x, 0x45670123u, 0xCDEF89ABu));  // xor 3
    x = fmaxf(x, pl16(x, 0x01234567u, 0x89ABCDEFu));  // xor 7
    x = fmaxf(x, pl16(x, 0xFEDCBA98u, 0x76543210u));  // xor 8
    return x;
}
__device__ __forceinline__ float rowsum16(float x) {
    x += pl16(x, 0x67452301u, 0xEFCDAB89u);
    x += pl16(x, 0x45670123u, 0xCDEF89ABu);
    x += pl16(x, 0x01234567u, 0x89ABCDEFu);
    x += pl16(x, 0xFEDCBA98u, 0x76543210u);
    return x;
}
#else
__device__ __forceinline__ float rowmax16(float x) {
    for (int d = 1; d < 16; d <<= 1) x = fmaxf(x, __shfl_xor(x, d));
    return x;
}
__device__ __forceinline__ float rowsum16(float x) {
    for (int d = 1; d < 16; d <<= 1) x += __shfl_xor(x, d);
    return x;
}
#endif

// ---------------------------------------------------------------------------
// f32 -> f16 conversion (vectorized)
// ---------------------------------------------------------------------------
__global__ void cvt_f32_f16(const float* __restrict__ in, _Float16* __restrict__ out, int n) {
    int i = blockIdx.x * blockDim.x + threadIdx.x;
    int stride = gridDim.x * blockDim.x;
    for (; i < (n >> 2); i += stride) {
        float4 f = ((const float4*)in)[i];
        union { _Float16 e[4]; uint2 u; } o;
        o.e[0] = (_Float16)f.x; o.e[1] = (_Float16)f.y;
        o.e[2] = (_Float16)f.z; o.e[3] = (_Float16)f.w;
        ((uint2*)out)[i] = o.u;
    }
}

// ---------------------------------------------------------------------------
// GEMM: Y[4096,1024] = A[4096,1024](f16) @ W[1024,1024]^T (f16)
// Block: 8 waves, 256M x 64N. Wave: 32M x 64N (2 A-frags x 4 acc).
// W tile (64x32) async-staged into LDS, double-buffered.
// ---------------------------------------------------------------------------
template<bool F32OUT>
__global__ void __launch_bounds__(256) gemm16(const _Float16* __restrict__ A,
                                              const _Float16* __restrict__ W,
                                              void* __restrict__ out) {
    __shared__ _Float16 wbuf[2][64][40];   // 40-halve stride: conflict-free b128 reads
    const int tid = threadIdx.x;
    const int lane = tid & 31, wv = tid >> 5;
    const int tn = blockIdx.x & 15;        // N/64 = 16
    const int tm = blockIdx.x >> 4;        // M/256 = 16
    const int hf = lane >> 4, lr = lane & 15;
    const int m0 = tm * 256 + wv * 32;
    const int n0 = tn * 64;
    const int scol = tid >> 2;             // 0..63 (column of W tile)
    const int sseg = (tid & 3) * 8;        // 16B segment within 32-k chunk

    copy16_async(&wbuf[0][scol][sseg], W + (size_t)(n0 + scol) * EE + sseg);

    v8f acc[2][4];
#pragma unroll
    for (int r = 0; r < 2; ++r)
#pragma unroll
        for (int j = 0; j < 4; ++j) acc[r][j] = (v8f){};

    for (int k0 = 0; k0 < EE; k0 += 32) {
        const int cb = (k0 >> 5) & 1;
        if (k0 + 32 < EE) {
            copy16_async(&wbuf[1 - cb][scol][sseg],
                         W + (size_t)(n0 + scol) * EE + k0 + 32 + sseg);
            WAIT_ASYNC(1);
        } else {
            WAIT_ASYNC(0);
        }
        __syncthreads();

        F16Frag a[2];
#pragma unroll
        for (int r = 0; r < 2; ++r) {
            const _Float16* ap = A + (size_t)(m0 + 16 * r + lr) * EE + k0;
#pragma unroll
            for (int i = 0; i < 8; ++i) a[r].u[i] = *(const uint32_t*)(ap + akb(i, hf));
        }
        if (k0 + 32 < EE)
            __builtin_prefetch(A + (size_t)(m0 + lr) * EE + k0 + 32, 0, 3);

#pragma unroll
        for (int j = 0; j < 4; ++j) {
            F16Frag bf;
            const _Float16* wp = &wbuf[cb][16 * j + lr][8 * hf];
            bf.q[0] = *(const uint4*)wp;
            bf.q[1] = *(const uint4*)(wp + 16);
#pragma unroll
            for (int r = 0; r < 2; ++r)
                acc[r][j] = __builtin_amdgcn_wmma_f32_16x16x32_f16(
                    false, a[r].h, false, bf.h, (short)0, acc[r][j], false, false);
        }
        __syncthreads();
    }
    // Epilogue: C/D layout -> element (vgpr v, lane): M = v + 8*hf, N = lane&15
#pragma unroll
    for (int r = 0; r < 2; ++r)
#pragma unroll
        for (int j = 0; j < 4; ++j)
#pragma unroll
            for (int v = 0; v < 8; ++v) {
                size_t m = (size_t)(m0 + 16 * r + v + 8 * hf);
                size_t n = (size_t)(n0 + 16 * j + lr);
                if (F32OUT) ((float*)out)[m * EE + n] = acc[r][j][v];
                else        ((_Float16*)out)[m * EE + n] = (_Float16)acc[r][j][v];
            }
}

// ---------------------------------------------------------------------------
// Quantum block: per (b,s,h) simulate 4-qubit circuit, overwrite first 4
// dk-elements of q/k/v (f16 buffers, layout [B*S, E] with head at h*DKK).
// ---------------------------------------------------------------------------
__global__ void quantum_inject(_Float16* qh, _Float16* kh, _Float16* vh,
                               const float* __restrict__ params) {
    int idx = blockIdx.x * blockDim.x + threadIdx.x;
    if (idx >= MROWS * HH) return;
    int row = idx >> 4;
    int h   = idx & 15;
    size_t base = (size_t)row * EE + (size_t)h * DKK;

    float g00x[8], g00y[8], g01x[8], g01y[8], g10x[8], g10y[8], g11x[8], g11y[8];
#pragma unroll
    for (int lq = 0; lq < 8; ++lq) {
        float phi = params[lq * 3 + 0], th = params[lq * 3 + 1], om = params[lq * 3 + 2];
        float ct = __cosf(th * 0.5f), sn = __sinf(th * 0.5f);
        float tp = 0.5f * (phi + om), tm = 0.5f * (phi - om);
        float cp = __cosf(tp), sp = __sinf(tp), cm = __cosf(tm), sm = __sinf(tm);
        g00x[lq] = ct * cp;  g00y[lq] = -ct * sp;
        g01x[lq] = -sn * cm; g01y[lq] = -sn * sm;
        g10x[lq] = sn * cm;  g10y[lq] = -sn * sm;
        g11x[lq] = ct * cp;  g11y[lq] = ct * sp;
    }

    _Float16* bufs[3] = { qh, kh, vh };
    for (int t = 0; t < 3; ++t) {
        _Float16* p = bufs[t] + base;
        float sx[16], sy[16];
#pragma unroll
        for (int i = 0; i < 16; ++i) { sx[i] = 0.f; sy[i] = 0.f; }
        sx[0] = 1.f;
#pragma unroll
        for (int q = 0; q < 4; ++q) {            // RX embedding, wire q = bit (3-q)
            float a = (float)p[q] * 0.5f;
            float c = __cosf(a), s = __sinf(a);
            int bit = 1 << (3 - q);
#pragma unroll
            for (int i = 0; i < 16; ++i) {
                if (i & bit) continue;
                int i1 = i | bit;
                float x0 = sx[i], y0 = sy[i], x1 = sx[i1], y1 = sy[i1];
                sx[i]  = c * x0 + s * y1;   sy[i]  = c * y0 - s * x1;
                sx[i1] = s * y0 + c * x1;   sy[i1] = -s * x0 + c * y1;
            }
        }
#pragma unroll
        for (int l = 0; l < 2; ++l) {
#pragma unroll
            for (int q = 0; q < 4; ++q) {        // Rot gates
                int lq = l * 4 + q;
                int bit = 1 << (3 - q);
#pragma unroll
                for (int i = 0; i < 16; ++i) {
                    if (i & bit) continue;
                    int i1 = i | bit;
                    float x0 = sx[i], y0 = sy[i], x1 = sx[i1], y1 = sy[i1];
                    float nx0 = g00x[lq]*x0 - g00y[lq]*y0 + g01x[lq]*x1 - g01y[lq]*y1;
                    float ny0 = g00x[lq]*y0 + g00y[lq]*x0 + g01x[lq]*y1 + g01y[lq]*x1;
                    float nx1 = g10x[lq]*x0 - g10y[lq]*y0 + g11x[lq]*x1 - g11y[lq]*y1;
                    float ny1 = g10x[lq]*y0 + g10y[lq]*x0 + g11x[lq]*y1 + g11y[lq]*x1;
                    sx[i] = nx0; sy[i] = ny0; sx[i1] = nx1; sy[i1] = ny1;
                }
            }
#pragma unroll
            for (int r = 0; r < 4; ++r) {        // CNOT(0,1): swap 8..11 <-> 12..15
                float tx = sx[8 + r], ty = sy[8 + r];
                sx[8 + r] = sx[12 + r]; sy[8 + r] = sy[12 + r];
                sx[12 + r] = tx; sy[12 + r] = ty;
            }
        }
#pragma unroll
        for (int q = 0; q < 4; ++q) {            // Pauli-Z expectations
            int bit = 1 << (3 - q);
            float ev = 0.f;
#pragma unroll
            for (int i = 0; i < 16; ++i) {
                float pr = sx[i] * sx[i] + sy[i] * sy[i];
                ev += (i & bit) ? -pr : pr;
            }
            p[q] = (_Float16)ev;
        }
    }
}

// ---------------------------------------------------------------------------
// Flash attention: block = 8 waves on one (b,h), 128 query rows.
// K/V 64-key tiles async-staged to LDS, double-buffered.
// ---------------------------------------------------------------------------
__global__ void __launch_bounds__(256) flash_attn(const _Float16* __restrict__ qh,
                                                  const _Float16* __restrict__ kh,
                                                  const _Float16* __restrict__ vh,
                                                  _Float16* __restrict__ oh) {
    __shared__ _Float16 kbuf[2][64][72];   // 72-halve stride: conflict-free b128 reads
    __shared__ _Float16 vbuf[2][64][72];
    __shared__ _Float16 pbuf[8][16][64];   // per-wave P staging

    const int tid = threadIdx.x;
    const int lane = tid & 31, wv = tid >> 5;
    const int qb = blockIdx.x & 15;        // S/128 = 16
    const int h  = (blockIdx.x >> 4) & 15;
    const int b  = blockIdx.x >> 8;
    const int hf = lane >> 4, lr = lane & 15;
    const int q0 = qb * 128 + wv * 16;
    const size_t rowbase = (size_t)(b * SS) * EE + (size_t)(h * DKK);

    const int sr = tid >> 3;               // 0..31 key row
    const int sc = (tid & 7) * 8;          // 16B dk segment

    // prefetch tile 0
    {
        const _Float16* kg = kh + rowbase + (size_t)sr * EE + sc;
        const _Float16* vg = vh + rowbase + (size_t)sr * EE + sc;
        copy16_async(&kbuf[0][sr][sc], kg);
        copy16_async(&vbuf[0][sr][sc], vg);
        copy16_async(&kbuf[0][sr + 32][sc], kg + (size_t)32 * EE);
        copy16_async(&vbuf[0][sr + 32][sc], vg + (size_t)32 * EE);
    }

    // Q fragments (dk 0..31, 32..63), pre-scaled by 1/sqrt(dk) = 1/8
    F16Frag qa[2];
#pragma unroll
    for (int t = 0; t < 2; ++t) {
        const _Float16* qp = qh + rowbase + (size_t)(q0 + lr) * EE + 32 * t;
#pragma unroll
        for (int i = 0; i < 8; ++i) {
            int kb = akb(i, hf);
            qa[t].e[2 * i]     = (_Float16)((float)qp[kb] * 0.125f);
            qa[t].e[2 * i + 1] = (_Float16)((float)qp[kb + 1] * 0.125f);
        }
    }

    float rm[8], rl[8];
#pragma unroll
    for (int v = 0; v < 8; ++v) { rm[v] = -1e30f; rl[v] = 0.f; }
    v8f o[4];
#pragma unroll
    for (int j = 0; j < 4; ++j) o[j] = (v8f){};

    for (int kt = 0; kt < SS; kt += 64) {
        const int cb = (kt >> 6) & 1;
        if (kt + 64 < SS) {
            const _Float16* kg = kh + rowbase + (size_t)(kt + 64 + sr) * EE + sc;
            const _Float16* vg = vh + rowbase + (size_t)(kt + 64 + sr) * EE + sc;
            copy16_async(&kbuf[1 - cb][sr][sc], kg);
            copy16_async(&vbuf[1 - cb][sr][sc], vg);
            copy16_async(&kbuf[1 - cb][sr + 32][sc], kg + (size_t)32 * EE);
            copy16_async(&vbuf[1 - cb][sr + 32][sc], vg + (size_t)32 * EE);
            WAIT_ASYNC(4);
        } else {
            WAIT_ASYNC(0);
        }
        __syncthreads();

        // --- scores: 4 sub-tiles of 16 keys, contraction over dk=64 ---
        v8f s[4];
#pragma unroll
        for (int t2 = 0; t2 < 4; ++t2) {
            v8f sc2 = (v8f){};
#pragma unroll
            for (int t = 0; t < 2; ++t) {
                F16Frag bf;
                const _Float16* kp = &kbuf[cb][16 * t2 + lr][32 * t + 8 * hf];
                bf.q[0] = *(const uint4*)kp;
                bf.q[1] = *(const uint4*)(kp + 16);
                sc2 = __builtin_amdgcn_wmma_f32_16x16x32_f16(
                    false, qa[t].h, false, bf.h, (short)0, sc2, false, false);
            }
            s[t2] = sc2;
        }

        // --- online softmax (rows: M = v + 8*hf; N across 16-lane half) ---
        float p[4][8];
#pragma unroll
        for (int v = 0; v < 8; ++v) {
            float mx = fmaxf(fmaxf(s[0][v], s[1][v]), fmaxf(s[2][v], s[3][v]));
            mx = rowmax16(mx);
            float mnew = fmaxf(rm[v], mx);
            float scale = __expf(rm[v] - mnew);
            float rs = 0.f;
#pragma unroll
            for (int t2 = 0; t2 < 4; ++t2) {
                p[t2][v] = __expf(s[t2][v] - mnew);
                rs += p[t2][v];
            }
            rs = rowsum16(rs);
            rl[v] = rl[v] * scale + rs;
            rm[v] = mnew;
#pragma unroll
            for (int j = 0; j < 4; ++j) o[j][v] *= scale;
        }

        // --- stage P (16 rows x 64 keys) through per-wave LDS ---
#pragma unroll
        for (int v = 0; v < 8; ++v) {
            int r = v + 8 * hf;
#pragma unroll
            for (int t2 = 0; t2 < 4; ++t2)
                pbuf[wv][r][16 * t2 + lr] = (_Float16)p[t2][v];
        }
        asm volatile("s_wait_dscnt 0x0" ::: "memory");
        F16Frag pa[2];
#pragma unroll
        for (int t = 0; t < 2; ++t)
#pragma unroll
            for (int i = 0; i < 8; ++i)
                pa[t].u[i] = *(const uint32_t*)&pbuf[wv][lr][32 * t + akb(i, hf)];

        // --- O += P @ V ---
#pragma unroll
        for (int j = 0; j < 4; ++j) {
            v8f oc = o[j];
#pragma unroll
            for (int t = 0; t < 2; ++t) {
                F16Frag vf;
#pragma unroll
                for (int i = 0; i < 8; ++i) {
                    int kk = 32 * t + akb(i, hf);
                    vf.e[2 * i]     = vbuf[cb][kk][16 * j + lr];
                    vf.e[2 * i + 1] = vbuf[cb][kk + 1][16 * j + lr];
                }
                oc = __builtin_amdgcn_wmma_f32_16x16x32_f16(
                    false, pa[t].h, false, vf.h, (short)0, oc, false, false);
            }
            o[j] = oc;
        }
        __syncthreads();
    }

    // --- epilogue: divide by row-sum, write f16 ---
#pragma unroll
    for (int j = 0; j < 4; ++j)
#pragma unroll
        for (int v = 0; v < 8; ++v) {
            size_t m = (size_t)(b * SS + q0 + v + 8 * hf);
            size_t d = (size_t)(h * DKK + 16 * j + lr);
            oh[m * EE + d] = (_Float16)(o[j][v] / rl[v]);
        }
}

// ---------------------------------------------------------------------------
extern "C" void kernel_launch(void* const* d_in, const int* in_sizes, int n_in,
                              void* d_out, int out_size, void* d_ws, size_t ws_size,
                              hipStream_t stream) {
    const float* x      = (const float*)d_in[0];
    const float* params = (const float*)d_in[1];
    const float* w_q    = (const float*)d_in[2];
    const float* w_k    = (const float*)d_in[3];
    const float* w_v    = (const float*)d_in[4];
    const float* w_o    = (const float*)d_in[5];
    float* out = (float*)d_out;

    const size_t MB = 1u << 20;
    char* ws = (char*)d_ws;
    _Float16* xh  = (_Float16*)(ws + 0 * MB);    // 4096x1024  (8 MB)
    _Float16* wqh = (_Float16*)(ws + 8 * MB);    // 1024x1024  (2 MB)
    _Float16* wkh = (_Float16*)(ws + 10 * MB);
    _Float16* wvh = (_Float16*)(ws + 12 * MB);
    _Float16* woh = (_Float16*)(ws + 14 * MB);
    _Float16* qhb = (_Float16*)(ws + 16 * MB);   // 8 MB each
    _Float16* khb = (_Float16*)(ws + 24 * MB);
    _Float16* vhb = (_Float16*)(ws + 32 * MB);
    _Float16* ohb = (_Float16*)(ws + 40 * MB);

    const int NX = MROWS * EE;
    const int NW = EE * EE;

    cvt_f32_f16<<<256, 256, 0, stream>>>(x,   xh,  NX);
    cvt_f32_f16<<<256, 256, 0, stream>>>(w_q, wqh, NW);
    cvt_f32_f16<<<256, 256, 0, stream>>>(w_k, wkh, NW);
    cvt_f32_f16<<<256, 256, 0, stream>>>(w_v, wvh, NW);
    cvt_f32_f16<<<256, 256, 0, stream>>>(w_o, woh, NW);

    // Q/K/V projections: 16 x 16 = 256 blocks (block = 256M x 64N)
    gemm16<false><<<256, 256, 0, stream>>>(xh, wqh, qhb);
    gemm16<false><<<256, 256, 0, stream>>>(xh, wkh, khb);
    gemm16<false><<<256, 256, 0, stream>>>(xh, wvh, vhb);

    // Quantum injection: B*S*H = 65536 threads
    quantum_inject<<<256, 256, 0, stream>>>(qhb, khb, vhb, params);

    // Flash attention: B*H*(S/128) = 512 blocks
    flash_attn<<<512, 256, 0, stream>>>(qhb, khb, vhb, ohb);

    // Output projection (f32 into d_out)
    gemm16<true><<<256, 256, 0, stream>>>(ohb, woh, out);
}
